// CausalSelfAttention_45019847197120
// MI455X (gfx1250) — matrix operand order repaired and checked by reading, not compiled, not run on hardware
//
#include <hip/hip_runtime.h>

// ---------------------------------------------------------------------------
// CDNA5 (gfx1250) causal self-attention.
// B=4, T=2048, C=2048, H=16, HD=128. bf16 WMMA (f32 accum) for all matmuls.
// GEMMs + flash-attention K-chunks double-buffered with
// GLOBAL_LOAD_ASYNC_TO_LDS (ASYNCcnt); ds_swizzle softmax reductions;
// exp2-domain online softmax.
// ---------------------------------------------------------------------------

#define B_NUM 4
#define T_SEQ 2048
#define C_DIM 2048
#define H_NUM 16
#define HD_DIM 128

typedef __attribute__((ext_vector_type(16))) __bf16 bf16x16;
typedef __attribute__((ext_vector_type(8)))  float  f32x8;
typedef __attribute__((ext_vector_type(4)))  float  f32x4;
typedef __attribute__((ext_vector_type(4)))  unsigned int u32x4;

// GCC-vector int4 in explicit address spaces, matching the async-LDS builtin's
// parameter types: (v4i addrspace(1)*, v4i addrspace(3)*, imm, imm)
typedef int v4i __attribute__((vector_size(16)));
typedef v4i __attribute__((address_space(1)))* gptr4;
typedef v4i __attribute__((address_space(3)))* lptr4;

#if defined(__has_builtin)
#if __has_builtin(__builtin_amdgcn_global_load_async_to_lds_b128)
#define USE_ASYNC 1
#endif
#endif
#ifndef USE_ASYNC
#define USE_ASYNC 0
#endif

#if USE_ASYNC
#if __has_builtin(__builtin_amdgcn_s_wait_asynccnt)
#define WAIT_ASYNC(n) __builtin_amdgcn_s_wait_asynccnt(n)
#else
#define WAIT_ASYNC(n) asm volatile("s_wait_asynccnt %0" ::"n"(n))
#endif
#define ASYNC_CP16(gsrc, ldst) \
  __builtin_amdgcn_global_load_async_to_lds_b128((gptr4)(gsrc), (lptr4)(ldst), 0, 0)
#else
#define WAIT_ASYNC(n)
#endif

// ds_swizzle XOR-mode lane exchange (EXEC all-ones at call sites):
// new_lane = lane ^ mask ; masks 1,2,4,8 stay within each 16-lane half.
#define SWZ_XOR(x, mask) \
  __int_as_float(__builtin_amdgcn_ds_swizzle(__float_as_int(x), (0x1F | ((mask) << 10))))

union AFrag { bf16x16 v; u32x4 q[2]; };
struct VRegs { u32x4 q[4]; };

__device__ __forceinline__ unsigned short f2bf(float f) {
  union { float f; unsigned int u; } c; c.f = f;
  unsigned int r = c.u + 0x7FFFu + ((c.u >> 16) & 1u);   // round-to-nearest-even
  return (unsigned short)(r >> 16);
}
__device__ __forceinline__ float bf2f(unsigned short h) {
  union { unsigned int u; float f; } c; c.u = ((unsigned int)h) << 16;
  return c.f;
}

// ---------------------------------------------------------------------------
// 1) f32 -> bf16 conversion, 4 elements/thread
// ---------------------------------------------------------------------------
__global__ __launch_bounds__(256) void cvt_f32_bf16_x4(const float* __restrict__ in,
                                                       unsigned short* __restrict__ out,
                                                       unsigned int n4) {
  unsigned int i = blockIdx.x * 256u + threadIdx.x;
  if (i >= n4) return;
  f32x4 f = *(const f32x4*)(in + (size_t)i * 4);
  union { unsigned long long q; unsigned short u[4]; } o;
#pragma unroll
  for (int p = 0; p < 4; ++p) o.u[p] = f2bf(f[p]);
  *(unsigned long long*)(out + (size_t)i * 4) = o.q;
}

// ---------------------------------------------------------------------------
// 2) GEMM: out[M,N] = A[M,K] * W[N,K]^T   (bf16 inputs, f32 accumulate)
//    Block tile 128x128, BK=32, 256 threads = 8 waves, wave tile 32x64.
//    Double-buffered LDS; async global->LDS copies when supported.
// ---------------------------------------------------------------------------
template <bool OUTF32>
__global__ __launch_bounds__(256) void gemm_xwT(const unsigned short* __restrict__ A,
                                                const unsigned short* __restrict__ W,
                                                void* __restrict__ outv,
                                                int M, int N, int K) {
  constexpr int AP = 40, BP = 40;             // row pitch in bf16 (80 B, 16B aligned)
  __shared__ __align__(16) unsigned short As[2][128 * AP];
  __shared__ __align__(16) unsigned short Bs[2][128 * BP];

  const int bn0 = blockIdx.x * 128;
  const int bm0 = blockIdx.y * 128;
  const int tid = threadIdx.x;
  const int lane = tid & 31;
  const int wave = __builtin_amdgcn_readfirstlane(tid >> 5);   // scalar wave id
  const int wm = (wave >> 1) * 32, wn = (wave & 1) * 64;
  const int lh = lane & 15, hi = lane >> 4;
  const int kbA = hi * 16;                    // A frag: 16 consecutive k at kbA
  const int kbB = hi * 16;                    // B frag: 16 consecutive k at kbB

  f32x8 acc[2][4] = {};

  auto loadTile = [&](int buf, int k0) {
#if USE_ASYNC
#pragma unroll
    for (int i = 0; i < 2; ++i) {
      int idx = tid + i * 256;
      int r = idx >> 2, c8 = idx & 3;
      ASYNC_CP16(A + (size_t)(bm0 + r) * K + k0 + c8 * 8,
                 &As[buf][r * AP + c8 * 8]);
    }
#pragma unroll
    for (int i = 0; i < 2; ++i) {
      int idx = tid + i * 256;
      int r = idx >> 2, c8 = idx & 3;
      ASYNC_CP16(W + (size_t)(bn0 + r) * K + k0 + c8 * 8,
                 &Bs[buf][r * BP + c8 * 8]);
    }
#else
#pragma unroll
    for (int i = 0; i < 2; ++i) {
      int idx = tid + i * 256;
      int r = idx >> 2, c8 = idx & 3;
      *(u32x4*)&As[buf][r * AP + c8 * 8] =
          *(const u32x4*)(A + (size_t)(bm0 + r) * K + k0 + c8 * 8);
    }
#pragma unroll
    for (int i = 0; i < 2; ++i) {
      int idx = tid + i * 256;
      int r = idx >> 2, c8 = idx & 3;
      *(u32x4*)&Bs[buf][r * BP + c8 * 8] =
          *(const u32x4*)(W + (size_t)(bn0 + r) * K + k0 + c8 * 8);
    }
#endif
  };

  const int nk = K >> 5;
  loadTile(0, 0);

  for (int t = 0; t < nk; ++t) {
    const int buf = t & 1;
    if (t + 1 < nk) {
      loadTile(buf ^ 1, (t + 1) * 32);
      WAIT_ASYNC(4);                          // previous tile's 4 async ops done
    } else {
      WAIT_ASYNC(0);
    }
    __syncthreads();

    AFrag a[2], bfr[4];
#pragma unroll
    for (int i = 0; i < 2; ++i) {
      const unsigned short* ap = &As[buf][(wm + i * 16 + lh) * AP + kbA];
      a[i].q[0] = *(const u32x4*)ap;
      a[i].q[1] = *(const u32x4*)(ap + 8);
    }
#pragma unroll
    for (int j = 0; j < 4; ++j) {
      const unsigned short* bp = &Bs[buf][(wn + j * 16 + lh) * BP + kbB];
      bfr[j].q[0] = *(const u32x4*)bp;
      bfr[j].q[1] = *(const u32x4*)(bp + 8);
    }
#pragma unroll
    for (int i = 0; i < 2; ++i)
#pragma unroll
      for (int j = 0; j < 4; ++j)
        acc[i][j] = __builtin_amdgcn_wmma_f32_16x16x32_bf16(
            false, a[i].v, false, bfr[j].v, (short)0, acc[i][j], false, false);
    __syncthreads();
  }

  // epilogue: C layout — VGPR v holds row (v + hi*8), lane lh holds col
#pragma unroll
  for (int i = 0; i < 2; ++i)
#pragma unroll
    for (int j = 0; j < 4; ++j)
#pragma unroll
      for (int v = 0; v < 8; ++v) {
        const size_t row = bm0 + wm + i * 16 + v + hi * 8;
        const size_t col = bn0 + wn + j * 16 + lh;
        if (OUTF32)
          ((float*)outv)[row * N + col] = acc[i][j][v];
        else
          ((unsigned short*)outv)[row * N + col] = f2bf(acc[i][j][v]);
      }
}

// ---------------------------------------------------------------------------
// 3) RoPE (interleaved pairs) applied in-place to bf16 Q and K, 4 pairs/thread
// ---------------------------------------------------------------------------
__global__ __launch_bounds__(256) void rope_qk(unsigned short* __restrict__ Q,
                                               unsigned short* __restrict__ Kt,
                                               const float* __restrict__ cosT,
                                               const float* __restrict__ sinT,
                                               unsigned int total4) {
  unsigned int idx = blockIdx.x * 256u + threadIdx.x;
  if (idx >= total4) return;                   // total4 = B*T*H*16
  const int g = idx & 15;                      // group of 4 pairs within head
  const unsigned int rem = idx >> 4;
  const int hh = rem & (H_NUM - 1);
  const unsigned int row = rem >> 4;           // b*T + t
  const int t = row & (T_SEQ - 1);
  const int i0 = g * 4;
  const f32x4 c = *(const f32x4*)(cosT + t * 64 + i0);
  const f32x4 s = *(const f32x4*)(sinT + t * 64 + i0);
  const size_t off = (size_t)row * C_DIM + hh * HD_DIM + 2 * i0;
  union { u32x4 q; unsigned short u[8]; } a, b;
  a.q = *(const u32x4*)(Q + off);
  b.q = *(const u32x4*)(Kt + off);
#pragma unroll
  for (int p = 0; p < 4; ++p) {
    float xr = bf2f(a.u[2 * p]), xi = bf2f(a.u[2 * p + 1]);
    a.u[2 * p]     = f2bf(xr * c[p] - xi * s[p]);
    a.u[2 * p + 1] = f2bf(xr * s[p] + xi * c[p]);
    xr = bf2f(b.u[2 * p]); xi = bf2f(b.u[2 * p + 1]);
    b.u[2 * p]     = f2bf(xr * c[p] - xi * s[p]);
    b.u[2 * p + 1] = f2bf(xr * s[p] + xi * c[p]);
  }
  *(u32x4*)(Q + off)  = a.q;
  *(u32x4*)(Kt + off) = b.q;
}

// ---------------------------------------------------------------------------
// 4) Flash attention: grid = (T/128, H, B), 256 threads = 8 waves,
//    each wave owns 16 q rows. Pipelined k-chunks: K double-buffered in LDS
//    via async copies; V prefetched to registers, scattered transposed next
//    iteration. Online softmax in exp2 domain; ds_swizzle reductions.
// ---------------------------------------------------------------------------
__global__ __launch_bounds__(256) void flash_attn(const unsigned short* __restrict__ Qg,
                                                  const unsigned short* __restrict__ Kg,
                                                  const unsigned short* __restrict__ Vg,
                                                  unsigned short* __restrict__ Og) {
  constexpr int KP = 136;   // K chunk pitch (bf16): 128 + 8 pad
  constexpr int VP = 72;    // V^T pitch: 64 keys + 8 pad
  constexpr int PP = 40;    // P tile pitch: 32 + 8 pad
  __shared__ __align__(16) unsigned short Ks[2][64 * KP];   // 2 x 17408 B
  __shared__ __align__(16) unsigned short Vt[128 * VP];     // 18432 B
  __shared__ __align__(16) unsigned short Pl[8 * 16 * PP];  // 10240 B

  const int qt = blockIdx.x, h = blockIdx.y, b = blockIdx.z;
  const int tid = threadIdx.x, lane = tid & 31;
  const int wave = __builtin_amdgcn_readfirstlane(tid >> 5); // scalar wave id
  const int lh = lane & 15, hi = lane >> 4;
  const int kbA = hi * 16;
  const size_t bh = (size_t)b * T_SEQ * C_DIM + (size_t)h * HD_DIM;
  const unsigned short* Qb = Qg + bh;
  const unsigned short* Kb = Kg + bh;
  const unsigned short* Vb = Vg + bh;
  unsigned short* Ob = (unsigned short*)Og + bh;

  const int kk0 = tid >> 4;                   // key row handled by this thread
  const int c8  = tid & 15;                   // dim/8 chunk handled

  auto issueK = [&](int kt, int buf) {        // async copy of 64x128 K chunk
#pragma unroll
    for (int i = 0; i < 4; ++i) {
      int kk = kk0 + i * 16;
      const size_t gro = (size_t)(kt * 64 + kk) * C_DIM + c8 * 8;
#if USE_ASYNC
      ASYNC_CP16(Kb + gro, &Ks[buf][kk * KP + c8 * 8]);
#else
      *(u32x4*)&Ks[buf][kk * KP + c8 * 8] = *(const u32x4*)(Kb + gro);
#endif
    }
  };
  auto loadV = [&](int kt) {                  // prefetch V chunk into registers
    VRegs r;
#pragma unroll
    for (int i = 0; i < 4; ++i) {
      int kk = kk0 + i * 16;
      r.q[i] = *(const u32x4*)(Vb + (size_t)(kt * 64 + kk) * C_DIM + c8 * 8);
    }
    return r;
  };

  // Q fragments (A role), kept in registers for whole kernel: 4 chunks over HD
  AFrag qa[4];
  {
    const unsigned short* qp = Qb + (size_t)(qt * 128 + wave * 16 + lh) * C_DIM;
#pragma unroll
    for (int c = 0; c < 4; ++c) {
      qa[c].q[0] = *(const u32x4*)(qp + c * 32 + kbA);
      qa[c].q[1] = *(const u32x4*)(qp + c * 32 + kbA + 8);
    }
  }

  f32x8 ov[8] = {};
  float mrow[8], lrow[8];
#pragma unroll
  for (int v = 0; v < 8; ++v) { mrow[v] = -1e30f; lrow[v] = 0.0f; }
  // exp2-domain: logits * (1/sqrt(HD)) * log2(e)
  const float qk_scale = 0.08838834764831845f * 1.4426950408889634f;
  const int qlo = qt * 128 + wave * 16;
  const int nkt = (qt + 1) * 2;               // causal: chunks of 64 keys
  unsigned short* Pw = &Pl[wave * 16 * PP];

  // pipeline prologue: chunk 0 in flight
  issueK(0, 0);
  VRegs vr = loadV(0), vrn;

  for (int kt = 0; kt < nkt; ++kt) {
    const int buf = kt & 1;
    // scatter chunk kt's V registers into transposed LDS (reads of the
    // previous chunk's Vt finished at the barrier ending last iteration)
#pragma unroll
    for (int i = 0; i < 4; ++i) {
      union { u32x4 q; unsigned short s[8]; } vt; vt.q = vr.q[i];
      int kk = kk0 + i * 16;
#pragma unroll
      for (int j = 0; j < 8; ++j) Vt[(c8 * 8 + j) * VP + kk] = vt.s[j];
    }
    if (kt + 1 < nkt) {
      issueK(kt + 1, buf ^ 1);
      WAIT_ASYNC(4);                          // chunk kt's 4 async ops done
      vrn = loadV(kt + 1);                    // overlap with compute below
    } else {
      WAIT_ASYNC(0);
    }
    __syncthreads();

#pragma unroll
    for (int jj = 0; jj < 2; ++jj) {
      const int kg = kt * 64 + jj * 32;
      if (kg <= qlo + 15) {                   // scalar skip of masked groups
        // S = Q K^T  (two 16x16 tiles, chained over HD in 32-chunks)
        f32x8 s[2] = {};
#pragma unroll
        for (int t2 = 0; t2 < 2; ++t2) {
          const int j = jj * 2 + t2;
#pragma unroll
          for (int c = 0; c < 4; ++c) {
            AFrag kf;
            const unsigned short* kp = &Ks[buf][(j * 16 + lh) * KP + c * 32 + hi * 16];
            kf.q[0] = *(const u32x4*)kp;
            kf.q[1] = *(const u32x4*)(kp + 8);
            s[t2] = __builtin_amdgcn_wmma_f32_16x16x32_bf16(
                false, qa[c].v, false, kf.v, (short)0, s[t2], false, false);
          }
        }
        // mask + online softmax in exp2 domain; 16-lane ds_swizzle reductions
#pragma unroll
        for (int v = 0; v < 8; ++v) {
          const int qrow = qlo + v + hi * 8;
          float v0 = s[0][v] * qk_scale, v1 = s[1][v] * qk_scale;
          if (kg + lh      > qrow) v0 = -1e30f;
          if (kg + 16 + lh > qrow) v1 = -1e30f;
          float tm = fmaxf(v0, v1);
          tm = fmaxf(tm, SWZ_XOR(tm, 1));
          tm = fmaxf(tm, SWZ_XOR(tm, 2));
          tm = fmaxf(tm, SWZ_XOR(tm, 4));
          tm = fmaxf(tm, SWZ_XOR(tm, 8));
          const float mnew = fmaxf(mrow[v], tm);
          const float fac = exp2f(mrow[v] - mnew);
          mrow[v] = mnew;
          const float p0 = exp2f(v0 - mnew), p1 = exp2f(v1 - mnew);
          float rs = p0 + p1;
          rs += SWZ_XOR(rs, 1);
          rs += SWZ_XOR(rs, 2);
          rs += SWZ_XOR(rs, 4);
          rs += SWZ_XOR(rs, 8);
          lrow[v] = lrow[v] * fac + rs;
#pragma unroll
          for (int d = 0; d < 8; ++d) ov[d][v] *= fac;
          // stash P (C layout -> LDS) for reload in A layout
          Pw[(v + hi * 8) * PP + lh]      = f2bf(p0);
          Pw[(v + hi * 8) * PP + 16 + lh] = f2bf(p1);
        }
        __builtin_amdgcn_wave_barrier();
        AFrag pa;
        const unsigned short* pp = &Pw[lh * PP + kbA];
        pa.q[0] = *(const u32x4*)pp;
        pa.q[1] = *(const u32x4*)(pp + 8);
        // O += P * V  (8 dim tiles of 16)
#pragma unroll
        for (int d = 0; d < 8; ++d) {
          AFrag vf;
          const unsigned short* vp = &Vt[(d * 16 + lh) * VP + jj * 32 + hi * 16];
          vf.q[0] = *(const u32x4*)vp;
          vf.q[1] = *(const u32x4*)(vp + 8);
          ov[d] = __builtin_amdgcn_wmma_f32_16x16x32_bf16(
              false, pa.v, false, vf.v, (short)0, ov[d], false, false);
        }
      }
    }
    __syncthreads();
    vr = vrn;
  }

  // normalize + store bf16 O
#pragma unroll
  for (int v = 0; v < 8; ++v) {
    const float inv = 1.0f / lrow[v];
    unsigned short* op = Ob + (size_t)(qlo + v + hi * 8) * C_DIM;
#pragma unroll
    for (int d = 0; d < 8; ++d) op[d * 16 + lh] = f2bf(ov[d][v] * inv);
  }
}

// ---------------------------------------------------------------------------
// Host launcher
// ---------------------------------------------------------------------------
extern "C" void kernel_launch(void* const* d_in, const int* in_sizes, int n_in,
                              void* d_out, int out_size, void* d_ws, size_t ws_size,
                              hipStream_t stream) {
  (void)in_sizes; (void)n_in; (void)out_size; (void)ws_size;
  const float* x    = (const float*)d_in[0];
  // d_in[1] = attn_mask (causality handled analytically)
  const float* cosT = (const float*)d_in[2];
  const float* sinT = (const float*)d_in[3];
  const float* Wq   = (const float*)d_in[4];
  const float* Wk   = (const float*)d_in[5];
  const float* Wv   = (const float*)d_in[6];
  const float* Wp   = (const float*)d_in[7];
  float* out = (float*)d_out;

  const size_t NX = (size_t)B_NUM * T_SEQ * C_DIM;   // 16M elements
  const size_t NW = (size_t)C_DIM * C_DIM;           // 4M elements
  unsigned short* wsb = (unsigned short*)d_ws;
  unsigned short* Xb  = wsb;
  unsigned short* Wqb = Xb  + NX;
  unsigned short* Wkb = Wqb + NW;
  unsigned short* Wvb = Wkb + NW;
  unsigned short* Wpb = Wvb + NW;
  unsigned short* Qb  = Wpb + NW;
  unsigned short* Kb  = Qb  + NX;
  unsigned short* Vb  = Kb  + NX;
  unsigned short* Ob  = Vb  + NX;

  // 1) convert inputs to bf16 (4 elements per thread)
  cvt_f32_bf16_x4<<<dim3((unsigned)(NX / 4 / 256)), 256, 0, stream>>>(x,  Xb,  (unsigned)(NX / 4));
  cvt_f32_bf16_x4<<<dim3((unsigned)(NW / 4 / 256)), 256, 0, stream>>>(Wq, Wqb, (unsigned)(NW / 4));
  cvt_f32_bf16_x4<<<dim3((unsigned)(NW / 4 / 256)), 256, 0, stream>>>(Wk, Wkb, (unsigned)(NW / 4));
  cvt_f32_bf16_x4<<<dim3((unsigned)(NW / 4 / 256)), 256, 0, stream>>>(Wv, Wvb, (unsigned)(NW / 4));
  cvt_f32_bf16_x4<<<dim3((unsigned)(NW / 4 / 256)), 256, 0, stream>>>(Wp, Wpb, (unsigned)(NW / 4));

  // 2) Q/K/V projections (bf16 out)
  const int M = B_NUM * T_SEQ;                       // 8192
  dim3 ggrid(C_DIM / 128, M / 128);                  // (16, 64)
  gemm_xwT<false><<<ggrid, 256, 0, stream>>>(Xb, Wqb, Qb, M, C_DIM, C_DIM);
  gemm_xwT<false><<<ggrid, 256, 0, stream>>>(Xb, Wkb, Kb, M, C_DIM, C_DIM);
  gemm_xwT<false><<<ggrid, 256, 0, stream>>>(Xb, Wvb, Vb, M, C_DIM, C_DIM);

  // 3) RoPE on Q and K (4 pairs per thread)
  const unsigned rtot4 = (unsigned)(B_NUM * T_SEQ * H_NUM * 16);
  rope_qk<<<dim3((rtot4 + 255) / 256), 256, 0, stream>>>(Qb, Kb, cosT, sinT, rtot4);

  // 4) flash attention -> bf16 O
  flash_attn<<<dim3(T_SEQ / 128, H_NUM, B_NUM), 256, 0, stream>>>(Qb, Kb, Vb, Ob);

  // 5) output projection (f32 out)
  gemm_xwT<true><<<ggrid, 256, 0, stream>>>(Ob, Wpb, out, M, C_DIM, C_DIM);
}
